// SlotAttention_71966472012123
// MI455X (gfx1250) — matrix operand order repaired and checked by reading, not compiled, not run on hardware
//
#include <hip/hip_runtime.h>
#include <hip/hip_bf16.h>
#include <math.h>

#define Bb 64
#define Nn 4096
#define Dd 256
#define Ss 8
#define Hh 512
#define ITERS 3
#define EPSF 1e-8f
#define LNEPS 1e-5f
#define UCH 32   // n-rows per TDM chunk in updates kernel

typedef __bf16 bf16x16 __attribute__((ext_vector_type(16)));
typedef float  f32x8   __attribute__((ext_vector_type(8)));
typedef unsigned int u32x4v __attribute__((ext_vector_type(4)));
typedef int i32x8v __attribute__((ext_vector_type(8)));
typedef int i32x4v __attribute__((ext_vector_type(4)));

union Frag16 { bf16x16 bf; uint4 q[2]; unsigned short us[16]; };

__device__ __forceinline__ unsigned short f2bf(float f) {
  unsigned u = __float_as_uint(f);
  u += 0x7fffu + ((u >> 16) & 1u);            // round-to-nearest-even
  return (unsigned short)(u >> 16);
}

// ---------------- utility kernels ----------------
__global__ void zero_u32_k(unsigned* p, long long n) {
  long long i = (long long)blockIdx.x * blockDim.x + threadIdx.x;
  if (i < n) p[i] = 0u;
}

__global__ void w2bf_k(const float* __restrict__ w, unsigned short* __restrict__ o, int n) {
  int i = blockIdx.x * blockDim.x + threadIdx.x;
  if (i < n) o[i] = f2bf(w[i]);
}

__global__ void init_slots_k(const float* __restrict__ noise, const float* __restrict__ mu,
                             const float* __restrict__ ls, float* __restrict__ slots) {
  int i = blockIdx.x * 256 + threadIdx.x;     // B*S*D threads
  int d = i & (Dd - 1);
  slots[i] = mu[d] + __expf(ls[d]) * noise[i];
}

__global__ void copy_k(const float* __restrict__ s, float* __restrict__ o) {
  int i = blockIdx.x * 256 + threadIdx.x;
  o[i] = s[i];
}

// ---------------- fused LayerNorm + K/V projection (bf16 WMMA) ----------------
// grid (N/64, B), 256 threads (8 waves). Each block: 64 input rows.
__global__ __launch_bounds__(256) void lnkv_k(
    const float* __restrict__ x, const float* __restrict__ lw, const float* __restrict__ lb,
    const unsigned short* __restrict__ wkb, const unsigned short* __restrict__ wvb,
    unsigned short* __restrict__ kbf, unsigned short* __restrict__ vbf)
{
  __shared__ unsigned short xb[64][264];
  __shared__ float ps[64][4], pq[64][4];
  __shared__ float mrow[64], rrow[64];
  const int tid = threadIdx.x;
  const int b = blockIdx.y, n0 = blockIdx.x * 64;
  const int r = tid & 63, qt = tid >> 6;
  const float* xrow = x + ((size_t)b * Nn + n0 + r) * Dd + qt * 64;
  float s = 0.f, sq = 0.f;
  for (int j = 0; j < 64; ++j) { float v = xrow[j]; s += v; sq += v * v; }
  ps[r][qt] = s; pq[r][qt] = sq;
  __syncthreads();
  if (tid < 64) {
    float s2 = ps[tid][0] + ps[tid][1] + ps[tid][2] + ps[tid][3];
    float q2 = pq[tid][0] + pq[tid][1] + pq[tid][2] + pq[tid][3];
    float m = s2 * (1.f / Dd);
    mrow[tid] = m;
    rrow[tid] = rsqrtf(q2 * (1.f / Dd) - m * m + LNEPS);
  }
  __syncthreads();
  {
    float m = mrow[r], rs = rrow[r];
    for (int j = 0; j < 64; ++j) {
      int col = qt * 64 + j;
      xb[r][col] = f2bf((xrow[j] - m) * rs * lw[col] + lb[col]);
    }
  }
  __syncthreads();
  // WMMA: 128 16x16 output tiles (k and v), 16 per wave, K=256.
  const int lane = tid & 31, w = tid >> 5;
  const int hi = lane >> 4, lm = lane & 15;
  for (int i = 0; i < 16; ++i) {
    int t = w * 16 + i;
    int mat = t >> 6, rem = t & 63;
    int mt = rem >> 4, dt = rem & 15;
    const unsigned short* wb = mat ? wvb : wkb;
    f32x8 acc = {};
    const unsigned short* xr = &xb[mt * 16 + lm][0];
    const unsigned short* wr = wb + (size_t)(dt * 16 + lm) * Dd;
#pragma unroll
    for (int j0 = 0; j0 < Dd; j0 += 32) {
      Frag16 af, bfr;
      af.q[0]  = *reinterpret_cast<const uint4*>(xr + j0 + hi * 16);
      af.q[1]  = *reinterpret_cast<const uint4*>(xr + j0 + hi * 16 + 8);
      bfr.q[0] = *reinterpret_cast<const uint4*>(wr + j0 + hi * 16);
      bfr.q[1] = *reinterpret_cast<const uint4*>(wr + j0 + hi * 16 + 8);
      acc = __builtin_amdgcn_wmma_f32_16x16x32_bf16(false, af.bf, false, bfr.bf,
                                                    (short)0, acc, false, false);
    }
    unsigned short* ob = (mat ? vbf : kbf) + ((size_t)b * Nn + n0 + mt * 16) * Dd + dt * 16 + lm;
#pragma unroll
    for (int rr = 0; rr < 8; ++rr) {
      int M = rr + 8 * hi;
      ob[(size_t)M * Dd] = f2bf(acc[rr]);
    }
  }
}

// ---------------- slots LN + q projection (VALU; tiny) ----------------
__global__ __launch_bounds__(256) void slotsq_k(
    const float* __restrict__ slots, const float* __restrict__ lw, const float* __restrict__ lb,
    const float* __restrict__ Wq, unsigned short* __restrict__ qbf, float* __restrict__ colsum)
{
  __shared__ float sn[8][256];
  __shared__ float ps[8][32], pq[8][32], mm[8], rr_[8];
  int tid = threadIdx.x, b = blockIdx.x;
  int r = tid >> 5, c = tid & 31;
  const float* srow = slots + ((size_t)b * Ss + r) * Dd;
  float s = 0.f, q = 0.f;
  for (int j = c; j < Dd; j += 32) { float v = srow[j]; s += v; q += v * v; }
  ps[r][c] = s; pq[r][c] = q;
  __syncthreads();
  if (tid < 8) {
    float s2 = 0.f, q2 = 0.f;
    for (int j = 0; j < 32; ++j) { s2 += ps[tid][j]; q2 += pq[tid][j]; }
    float m = s2 * (1.f / Dd);
    mm[tid] = m;
    rr_[tid] = rsqrtf(q2 * (1.f / Dd) - m * m + LNEPS);
  }
  __syncthreads();
  {
    float m = mm[r], rs = rr_[r];
    for (int j = c; j < Dd; j += 32) sn[r][j] = (srow[j] - m) * rs * lw[j] + lb[j];
  }
  if (tid < 8) colsum[b * 8 + tid] = 0.f;   // re-zero for this iteration's attn pass
  __syncthreads();
  float acc[8] = {};
  const float* wrow = Wq + (size_t)tid * Dd;
  for (int j = 0; j < Dd; ++j) {
    float wv = wrow[j];
#pragma unroll
    for (int s2 = 0; s2 < 8; ++s2) acc[s2] += sn[s2][j] * wv;
  }
  const float scale = 0.0625f;              // 256^-0.5
  for (int s2 = 0; s2 < 8; ++s2)
    qbf[((size_t)b * 16 + s2) * Dd + tid] = f2bf(acc[s2] * scale);
}

// ---------------- logits (WMMA) + softmax + transpose-store ----------------
// grid (N/128, B), 256 threads (8 waves -> 8 row-tiles of 16).
__global__ __launch_bounds__(256) void attn_k(
    const unsigned short* __restrict__ kbf, const unsigned short* __restrict__ qbf,
    unsigned short* __restrict__ attn_t, float* __restrict__ colsum)
{
  __shared__ float lg[128][8];
  __shared__ float cs[8];
  const int tid = threadIdx.x;
  const int b = blockIdx.y, n0 = blockIdx.x * 128;
  if (tid < 8) cs[tid] = 0.f;
  const int lane = tid & 31, w = tid >> 5;
  const int hi = lane >> 4, lm = lane & 15;
  const unsigned short* kr = kbf + ((size_t)b * Nn + n0 + w * 16 + lm) * Dd;
  const unsigned short* qr = qbf + ((size_t)b * 16 + lm) * Dd;
  f32x8 acc = {};
#pragma unroll
  for (int j0 = 0; j0 < Dd; j0 += 32) {
    Frag16 af, bfr;
    af.q[0]  = *reinterpret_cast<const uint4*>(kr + j0 + hi * 16);
    af.q[1]  = *reinterpret_cast<const uint4*>(kr + j0 + hi * 16 + 8);
    bfr.q[0] = *reinterpret_cast<const uint4*>(qr + j0 + hi * 16);
    bfr.q[1] = *reinterpret_cast<const uint4*>(qr + j0 + hi * 16 + 8);
    acc = __builtin_amdgcn_wmma_f32_16x16x32_bf16(false, af.bf, false, bfr.bf,
                                                  (short)0, acc, false, false);
  }
  if (lm < 8) {
#pragma unroll
    for (int rr = 0; rr < 8; ++rr) lg[w * 16 + rr + 8 * hi][lm] = acc[rr];
  }
  __syncthreads();
  if (tid < 128) {
    float v[8]; float mx = -1e30f;
    for (int s2 = 0; s2 < 8; ++s2) { v[s2] = lg[tid][s2]; mx = fmaxf(mx, v[s2]); }
    float sum = 0.f;
    for (int s2 = 0; s2 < 8; ++s2) { v[s2] = __expf(v[s2] - mx); sum += v[s2]; }
    float inv = 1.f / sum;
    unsigned short* ap = attn_t + ((size_t)b * 16) * Nn + n0 + tid;
    for (int s2 = 0; s2 < 8; ++s2) {
      float a = v[s2] * inv + EPSF;
      ap[(size_t)s2 * Nn] = f2bf(a);
      atomicAdd(&cs[s2], a);
    }
  }
  __syncthreads();
  if (tid < 8) atomicAdd(&colsum[b * 8 + tid], cs[tid]);
}

// ---------------- updates = (attn^T @ v) / colsum  (WMMA + TDM staging) ----------------
#if __has_builtin(__builtin_amdgcn_tensor_load_to_lds)
#define HAVE_TDM 1
__device__ __forceinline__ void tdm_issue(const unsigned short* gptr, unsigned lds_off) {
  unsigned long long ga = (unsigned long long)gptr;
  u32x4v g0;
  g0[0] = 1u;                                          // count=1, user descriptor
  g0[1] = lds_off;                                     // lds_addr
  g0[2] = (unsigned)ga;                                // global_addr[31:0]
  g0[3] = (unsigned)((ga >> 32) & 0x1FFFFFFu) | (2u << 30);  // global_addr[56:32] | type=2
  i32x8v g1;
  g1[0] = (int)(1u << 16);                             // workgroup_mask=0, data_size=2B
  g1[1] = (int)(((unsigned)Dd & 0xffffu) << 16);       // tensor_dim0 lo16 = 256
  g1[2] = (int)(((unsigned)Nn & 0xffffu) << 16);       // dim0 hi=0 | tensor_dim1 lo16 = 4096
  g1[3] = (int)(((unsigned)Dd) << 16);                 // dim1 hi=0 | tile_dim0 = 256
  g1[4] = (int)UCH;                                    // tile_dim1 = 32 | tile_dim2 = 0
  g1[5] = (int)Dd;                                     // tensor_dim0_stride lo32 = 256
  g1[6] = 0;
  g1[7] = 0;
  i32x4v z4 = {0, 0, 0, 0};
  i32x8v z8 = {0, 0, 0, 0, 0, 0, 0, 0};
  // 6-arg form (this toolchain): (g0, g1, g2, g3, extra_group, cpol)
  __builtin_amdgcn_tensor_load_to_lds(g0, g1, z4, z4, z8, 0);
}
#endif

// grid (B), 512 threads (16 waves, one 16-wide d-tile each).
__global__ __launch_bounds__(512) void updates_k(
    const unsigned short* __restrict__ attn_t, const unsigned short* __restrict__ vbf,
    const float* __restrict__ colsum, float* __restrict__ upd)
{
  __shared__ unsigned short vt[2][UCH * Dd];
  int tid = threadIdx.x, b = blockIdx.x;
  int lane = tid & 31, w = tid >> 5;
  int hi = lane >> 4, lm = lane & 15;
  const int nch = Nn / UCH;                            // 128 chunks
  const unsigned short* arow = attn_t + ((size_t)b * 16 + lm) * Nn;
  const int dcol = w * 16 + lm;
  f32x8 acc = {};
#ifdef HAVE_TDM
  unsigned ldsbase[2];
  {
    __attribute__((address_space(3))) unsigned short* p0 =
        (__attribute__((address_space(3))) unsigned short*)&vt[0][0];
    __attribute__((address_space(3))) unsigned short* p1 =
        (__attribute__((address_space(3))) unsigned short*)&vt[1][0];
    ldsbase[0] = (unsigned)(unsigned long long)p0;
    ldsbase[1] = (unsigned)(unsigned long long)p1;
  }
  if (w == 0) tdm_issue(vbf + ((size_t)b * Nn) * Dd, ldsbase[0]);
  for (int c = 0; c < nch; ++c) {
    if (w == 0) {
      if (c + 1 < nch) {
        tdm_issue(vbf + ((size_t)b * Nn + (size_t)(c + 1) * UCH) * Dd, ldsbase[(c + 1) & 1]);
        __builtin_amdgcn_s_wait_tensorcnt(1);          // chunk c complete (in-order)
      } else {
        __builtin_amdgcn_s_wait_tensorcnt(0);
      }
    }
    __syncthreads();                                   // chunk c visible to all waves
    {
      int nb = c * UCH;
      Frag16 af, bfr;
      af.q[0] = *reinterpret_cast<const uint4*>(arow + nb + hi * 16);
      af.q[1] = *reinterpret_cast<const uint4*>(arow + nb + hi * 16 + 8);
      const unsigned short* vloc = &vt[c & 1][0];
#pragma unroll
      for (int e = 0; e < 16; ++e) bfr.us[e] = vloc[(hi * 16 + e) * Dd + dcol];
      acc = __builtin_amdgcn_wmma_f32_16x16x32_bf16(false, af.bf, false, bfr.bf,
                                                    (short)0, acc, false, false);
    }
    __syncthreads();                                   // done reading before buffer reuse
  }
#else
  for (int c = 0; c < nch; ++c) {
    const unsigned short* vsrc = vbf + ((size_t)b * Nn + (size_t)c * UCH) * Dd;
    __syncthreads();
    for (int i = tid; i < UCH * Dd; i += 512) vt[0][i] = vsrc[i];
    __syncthreads();
    int nb = c * UCH;
    Frag16 af, bfr;
    af.q[0] = *reinterpret_cast<const uint4*>(arow + nb + hi * 16);
    af.q[1] = *reinterpret_cast<const uint4*>(arow + nb + hi * 16 + 8);
#pragma unroll
    for (int e = 0; e < 16; ++e) bfr.us[e] = vt[0][(hi * 16 + e) * Dd + dcol];
    acc = __builtin_amdgcn_wmma_f32_16x16x32_bf16(false, af.bf, false, bfr.bf,
                                                  (short)0, acc, false, false);
  }
#endif
  if (hi == 0) {                                       // rows M=0..7 == slots 0..7
#pragma unroll
    for (int rr = 0; rr < 8; ++rr) {
      float inv = 1.f / colsum[b * 8 + rr];
      upd[((size_t)b * Ss + rr) * Dd + dcol] = acc[rr] * inv;
    }
  }
}

// ---------------- GRU cell (VALU; weight rows reused across 8 slots) ----------------
__global__ __launch_bounds__(256) void gru_k(
    const float* __restrict__ upd, float* __restrict__ slots,
    const float* __restrict__ wih, const float* __restrict__ whh,
    const float* __restrict__ bih, const float* __restrict__ bhh)
{
  __shared__ float xs[8][256], hsh[8][256];
  int tid = threadIdx.x, b = blockIdx.x;
  for (int i = tid; i < Ss * Dd; i += 256) {
    xs[i >> 8][i & 255]  = upd[(size_t)b * Ss * Dd + i];
    hsh[i >> 8][i & 255] = slots[(size_t)b * Ss * Dd + i];
  }
  __syncthreads();
  float ar[8] = {}, az[8] = {}, an[8] = {}, hr[8] = {}, hz[8] = {}, hn[8] = {};
  const float* wi_r = wih + (size_t)tid * Dd;
  const float* wi_z = wih + (size_t)(Dd + tid) * Dd;
  const float* wi_n = wih + (size_t)(2 * Dd + tid) * Dd;
  const float* wh_r = whh + (size_t)tid * Dd;
  const float* wh_z = whh + (size_t)(Dd + tid) * Dd;
  const float* wh_n = whh + (size_t)(2 * Dd + tid) * Dd;
  for (int j = 0; j < Dd; ++j) {
    float wr = wi_r[j], wz = wi_z[j], wn = wi_n[j];
    float vr = wh_r[j], vz = wh_z[j], vn = wh_n[j];
#pragma unroll
    for (int s2 = 0; s2 < 8; ++s2) {
      float xv = xs[s2][j], hv = hsh[s2][j];
      ar[s2] += wr * xv; az[s2] += wz * xv; an[s2] += wn * xv;
      hr[s2] += vr * hv; hz[s2] += vz * hv; hn[s2] += vn * hv;
    }
  }
  float bir = bih[tid], biz = bih[Dd + tid], bin_ = bih[2 * Dd + tid];
  float bhr = bhh[tid], bhz = bhh[Dd + tid], bhn = bhh[2 * Dd + tid];
  for (int s2 = 0; s2 < 8; ++s2) {
    float r = 1.f / (1.f + __expf(-(ar[s2] + bir + hr[s2] + bhr)));
    float z = 1.f / (1.f + __expf(-(az[s2] + biz + hz[s2] + bhz)));
    float n = tanhf(an[s2] + bin_ + r * (hn[s2] + bhn));
    slots[(size_t)b * Ss * Dd + s2 * Dd + tid] = (1.f - z) * n + z * hsh[s2][tid];
  }
}

// ---------------- residual MLP (VALU) ----------------
__global__ __launch_bounds__(256) void mlp_k(
    float* __restrict__ slots, const float* __restrict__ lw, const float* __restrict__ lb,
    const float* __restrict__ w1, const float* __restrict__ b1,
    const float* __restrict__ w2, const float* __restrict__ b2)
{
  __shared__ float sn[8][256];
  __shared__ float hl[8][512];
  __shared__ float ps[8][32], pq[8][32], mm[8], rr_[8];
  int tid = threadIdx.x, b = blockIdx.x;
  int r = tid >> 5, c = tid & 31;
  const float* srow = slots + ((size_t)b * Ss + r) * Dd;
  float s = 0.f, q = 0.f;
  for (int j = c; j < Dd; j += 32) { float v = srow[j]; s += v; q += v * v; }
  ps[r][c] = s; pq[r][c] = q;
  __syncthreads();
  if (tid < 8) {
    float s2 = 0.f, q2 = 0.f;
    for (int j = 0; j < 32; ++j) { s2 += ps[tid][j]; q2 += pq[tid][j]; }
    float m = s2 * (1.f / Dd);
    mm[tid] = m;
    rr_[tid] = rsqrtf(q2 * (1.f / Dd) - m * m + LNEPS);
  }
  __syncthreads();
  {
    float m = mm[r], rs = rr_[r];
    for (int j = c; j < Dd; j += 32) sn[r][j] = (srow[j] - m) * rs * lw[j] + lb[j];
  }
  __syncthreads();
  float a1[8] = {}, a2[8] = {};
  const float* w1a = w1 + (size_t)tid * Dd;
  const float* w1b = w1 + (size_t)(tid + 256) * Dd;
  for (int j = 0; j < Dd; ++j) {
    float u = w1a[j], v = w1b[j];
#pragma unroll
    for (int s2 = 0; s2 < 8; ++s2) { float x = sn[s2][j]; a1[s2] += u * x; a2[s2] += v * x; }
  }
  float bb1 = b1[tid], bb2 = b1[tid + 256];
  for (int s2 = 0; s2 < 8; ++s2) {
    hl[s2][tid]       = fmaxf(a1[s2] + bb1, 0.f);
    hl[s2][tid + 256] = fmaxf(a2[s2] + bb2, 0.f);
  }
  __syncthreads();
  float o[8] = {};
  const float* w2r = w2 + (size_t)tid * Hh;
  for (int j = 0; j < Hh; ++j) {
    float u = w2r[j];
#pragma unroll
    for (int s2 = 0; s2 < 8; ++s2) o[s2] += hl[s2][j] * u;
  }
  float bo = b2[tid];
  for (int s2 = 0; s2 < 8; ++s2)
    slots[((size_t)b * Ss + s2) * Dd + tid] += o[s2] + bo;
}

// ---------------- host launcher ----------------
extern "C" void kernel_launch(void* const* d_in, const int* in_sizes, int n_in,
                              void* d_out, int out_size, void* d_ws, size_t ws_size,
                              hipStream_t stream)
{
  (void)in_sizes; (void)n_in; (void)out_size; (void)ws_size;
  const float* inputs = (const float*)d_in[0];
  const float* noise  = (const float*)d_in[1];
  const float* ln_in_w = (const float*)d_in[2];
  const float* ln_in_b = (const float*)d_in[3];
  const float* ln_sl_w = (const float*)d_in[4];
  const float* ln_sl_b = (const float*)d_in[5];
  const float* ln_ml_w = (const float*)d_in[6];
  const float* ln_ml_b = (const float*)d_in[7];
  const float* mu   = (const float*)d_in[8];
  const float* lsig = (const float*)d_in[9];
  const float* Wq = (const float*)d_in[10];
  const float* Wk = (const float*)d_in[11];
  const float* Wv = (const float*)d_in[12];
  const float* gwih = (const float*)d_in[13];
  const float* gwhh = (const float*)d_in[14];
  const float* gbih = (const float*)d_in[15];
  const float* gbhh = (const float*)d_in[16];
  const float* w1 = (const float*)d_in[17];
  const float* b1 = (const float*)d_in[18];
  const float* w2 = (const float*)d_in[19];
  const float* b2 = (const float*)d_in[20];

  char* ws = (char*)d_ws;
  size_t off = 0;
  auto alloc = [&](size_t bytes) { size_t o = off; off += (bytes + 255) & ~(size_t)255; return o; };
  unsigned short* kbf = (unsigned short*)(ws + alloc((size_t)Bb * Nn * Dd * 2));
  unsigned short* vbf = (unsigned short*)(ws + alloc((size_t)Bb * Nn * Dd * 2));
  unsigned short* qbf = (unsigned short*)(ws + alloc((size_t)Bb * 16 * Dd * 2));
  unsigned short* at  = (unsigned short*)(ws + alloc((size_t)Bb * 16 * Nn * 2));
  float* slots  = (float*)(ws + alloc((size_t)Bb * Ss * Dd * 4));
  float* upd    = (float*)(ws + alloc((size_t)Bb * Ss * Dd * 4));
  float* colsum = (float*)(ws + alloc((size_t)Bb * Ss * 4));
  unsigned short* wkb = (unsigned short*)(ws + alloc((size_t)Dd * Dd * 2));
  unsigned short* wvb = (unsigned short*)(ws + alloc((size_t)Dd * Dd * 2));

  // one-time per call: weight conversion, pad-row zeroing, slot init, K/V projection
  w2bf_k<<<dim3((Dd * Dd + 255) / 256), 256, 0, stream>>>(Wk, wkb, Dd * Dd);
  w2bf_k<<<dim3((Dd * Dd + 255) / 256), 256, 0, stream>>>(Wv, wvb, Dd * Dd);
  long long zc1 = (long long)Bb * 16 * Nn * 2 / 4;
  zero_u32_k<<<dim3((unsigned)((zc1 + 255) / 256)), 256, 0, stream>>>((unsigned*)at, zc1);
  long long zc2 = (long long)Bb * 16 * Dd * 2 / 4;
  zero_u32_k<<<dim3((unsigned)((zc2 + 255) / 256)), 256, 0, stream>>>((unsigned*)qbf, zc2);
  init_slots_k<<<dim3(Bb * Ss * Dd / 256), 256, 0, stream>>>(noise, mu, lsig, slots);
  lnkv_k<<<dim3(Nn / 64, Bb), 256, 0, stream>>>(inputs, ln_in_w, ln_in_b, wkb, wvb, kbf, vbf);

  for (int it = 0; it < ITERS; ++it) {
    slotsq_k<<<dim3(Bb), 256, 0, stream>>>(slots, ln_sl_w, ln_sl_b, Wq, qbf, colsum);
    attn_k<<<dim3(Nn / 128, Bb), 256, 0, stream>>>(kbf, qbf, at, colsum);
    updates_k<<<dim3(Bb), 512, 0, stream>>>(at, vbf, colsum, upd);
    gru_k<<<dim3(Bb), 256, 0, stream>>>(upd, slots, gwih, gwhh, gbih, gbhh);
    mlp_k<<<dim3(Bb), 256, 0, stream>>>(slots, ln_ml_w, ln_ml_b, w1, b1, w2, b2);
  }
  copy_k<<<dim3(Bb * Ss * Dd / 256), 256, 0, stream>>>(slots, (float*)d_out);
}
